// NanoGPT_28028956574150
// MI455X (gfx1250) — matrix operand order repaired and checked by reading, not compiled, not run on hardware
//
#include <hip/hip_runtime.h>
#include <hip/hip_bf16.h>
#include <math.h>

typedef __bf16 bf16;
typedef __attribute__((ext_vector_type(16))) __bf16 v16bf;
typedef __attribute__((ext_vector_type(8)))  float  v8f;
typedef __attribute__((ext_vector_type(4)))  unsigned int u32x4;

// ---------------------------------------------------------------- constants
constexpr int Bc  = 4;
constexpr int Tc  = 1024;
constexpr int Dc  = 768;
constexpr int Lc  = 12;
constexpr int Hc  = 12;
constexpr int HDc = 64;
constexpr int Vc  = 50257;
constexpr int Mc  = Bc * Tc;      // 4096 rows of the token stream
constexpr int N3c = 3 * Hc * HDc; // 2304
constexpr int Fc  = 4 * Dc;       // 3072

// ---------------------------------------------------------------- helpers
struct Frag32 { u32x4 lo, hi; };

static __device__ __forceinline__ v16bf make_frag(u32x4 lo, u32x4 hi) {
    Frag32 t; t.lo = lo; t.hi = hi;
    return __builtin_bit_cast(v16bf, t);
}

#define WMMA_BF16(a, b, c) \
    __builtin_amdgcn_wmma_f32_16x16x32_bf16(false, (a), false, (b), (short)0, (c), false, false)

// ---------------------------------------------------------------- f32 -> bf16
__global__ void cvt_kernel(const float* __restrict__ in, bf16* __restrict__ out, size_t n) {
    size_t i = (size_t)blockIdx.x * blockDim.x + threadIdx.x;
    size_t stride = (size_t)gridDim.x * blockDim.x;
    for (; i < n; i += stride) out[i] = (bf16)in[i];
}

// ---------------------------------------------------------------- embedding
__global__ void embed_kernel(const int* __restrict__ ids,
                             const float* __restrict__ tok,
                             const float* __restrict__ pos,
                             float* __restrict__ h) {
    size_t n = (size_t)Mc * Dc;
    size_t i = (size_t)blockIdx.x * blockDim.x + threadIdx.x;
    size_t stride = (size_t)gridDim.x * blockDim.x;
    for (; i < n; i += stride) {
        size_t bt = i / Dc;
        int d = (int)(i % Dc);
        int t = (int)(bt % Tc);
        int tk = ids[bt];
        h[i] = tok[(size_t)tk * Dc + d] + pos[(size_t)t * Dc + d];
    }
}

// ---------------------------------------------------------------- layernorm -> bf16
__global__ __launch_bounds__(256) void layernorm_kernel(const float* __restrict__ x,
                                                        const float* __restrict__ g,
                                                        const float* __restrict__ b,
                                                        bf16* __restrict__ out, int D) {
    __shared__ float red[256];
    __shared__ float red2[256];
    const int row = blockIdx.x, tid = threadIdx.x;
    const float* xr = x + (size_t)row * D;
    float s = 0.f, s2 = 0.f;
    for (int d = tid; d < D; d += 256) { float v = xr[d]; s += v; s2 += v * v; }
    red[tid] = s; red2[tid] = s2; __syncthreads();
    for (int off = 128; off > 0; off >>= 1) {
        if (tid < off) { red[tid] += red[tid + off]; red2[tid] += red2[tid + off]; }
        __syncthreads();
    }
    const float mu  = red[0] / D;
    const float var = red2[0] / D - mu * mu;
    const float rs  = rsqrtf(var + 1e-5f);
    bf16* orow = out + (size_t)row * D;
    for (int d = tid; d < D; d += 256)
        orow[d] = (bf16)((xr[d] - mu) * rs * g[d] + b[d]);
}

// ---------------------------------------------------------------- bf16 WMMA GEMM
// C[M,N] = A[M,K] @ B[K,N] (+bias) (relu) (+resid) -> outF (f32) and/or outB (bf16)
// block: 256 thr (8 waves); tile 128x64x32; wave tile 32x32 = 2x2 WMMA 16x16x32
template <bool NSAFE>
__global__ __launch_bounds__(256) void gemm_bf16_kernel(const bf16* __restrict__ A,
                                                        const bf16* __restrict__ Bw,
                                                        const float* __restrict__ bias,
                                                        const float* __restrict__ resid,
                                                        float* __restrict__ outF,
                                                        bf16* __restrict__ outB,
                                                        int M, int N, int K, int relu) {
    __shared__ __align__(16) bf16 As[128 * 32];
    __shared__ __align__(16) bf16 Bs[32 * 64];

    const int tid  = threadIdx.x;
    const int lane = tid & 31;
    // wave index is uniform across the wave: force it into an SGPR so all
    // wave-tile math is SALU and control flow never touches EXEC.
    const int wid  = __builtin_amdgcn_readfirstlane(tid >> 5);
    const int wm   = wid >> 1, wn = wid & 1;
    const int gm0  = blockIdx.y * 128;
    const int gn0  = blockIdx.x * 64;

    v8f acc[2][2] = {};

    const int mlane = lane & 15;
    const int khalf = lane >> 4;

    const int ksteps = K >> 5;
    for (int kt = 0; kt < ksteps; ++kt) {
        const int k0 = kt << 5;
        // stage A tile 128x32 (2 x 16B per thread)
#pragma unroll
        for (int c = 0; c < 2; ++c) {
            int id  = tid * 2 + c;
            int e   = id * 8;
            int row = e >> 5, col = e & 31;
            u32x4 v = *(const u32x4*)(A + (size_t)(gm0 + row) * K + k0 + col);
            *(u32x4*)(As + row * 32 + col) = v;
        }
        // stage B tile 32x64 (1 x 16B per thread, or scalar+guard for odd N)
        {
            int e   = tid * 8;
            int row = e >> 6, col = e & 63;
            if (NSAFE) {
                u32x4 v = *(const u32x4*)(Bw + (size_t)(k0 + row) * N + gn0 + col);
                *(u32x4*)(Bs + row * 64 + col) = v;
            } else {
#pragma unroll
                for (int i = 0; i < 8; ++i) {
                    int gc = gn0 + col + i;
                    Bs[row * 64 + col + i] = (gc < N) ? Bw[(size_t)(k0 + row) * N + gc]
                                                      : (bf16)0.0f;
                }
            }
        }
        __syncthreads();

        v16bf af[2], bfr[2];
#pragma unroll
        for (int i = 0; i < 2; ++i) {
            int m = wm * 32 + i * 16 + mlane;
            const u32x4* lo = (const u32x4*)(As + m * 32 + khalf * 8);
            const u32x4* hi = (const u32x4*)(As + m * 32 + 16 + khalf * 8);
            af[i] = make_frag(*lo, *hi);
        }
#pragma unroll
        for (int j = 0; j < 2; ++j) {
            int nb = wn * 32 + j * 16;
#pragma unroll
            for (int s = 0; s < 16; ++s) {
                int k = (s < 8) ? (khalf * 8 + s) : (16 + khalf * 8 + (s - 8));
                bfr[j][s] = Bs[k * 64 + nb + mlane];
            }
        }
#pragma unroll
        for (int i = 0; i < 2; ++i)
#pragma unroll
            for (int j = 0; j < 2; ++j)
                acc[i][j] = WMMA_BF16(af[i], bfr[j], acc[i][j]);
        __syncthreads();
    }

    // store: lane = N col, VGPR r = M row (halves split by lane/16)
    const int nlane = lane & 15, mhalf = lane >> 4;
#pragma unroll
    for (int i = 0; i < 2; ++i) {
#pragma unroll
        for (int j = 0; j < 2; ++j) {
            int gcol = gn0 + wn * 32 + j * 16 + nlane;
            if (!NSAFE && gcol >= N) continue;
            float bv = bias ? bias[gcol] : 0.0f;
#pragma unroll
            for (int r = 0; r < 8; ++r) {
                int grow = gm0 + wm * 32 + i * 16 + mhalf * 8 + r;
                float v  = acc[i][j][r] + bv;
                if (relu) v = fmaxf(v, 0.0f);
                if (resid) v += resid[(size_t)grow * N + gcol];
                if (outF) outF[(size_t)grow * N + gcol] = v;
                if (outB) outB[(size_t)grow * N + gcol] = (bf16)v;
            }
        }
    }
}

// ---------------------------------------------------------------- qkv repack
// qkv[B*T, 2304] (order per head: k | q | v) -> Q[B,H,T,HD]*scale, K^T[B,H,HD,T], V[B,H,T,HD]
__global__ void repack_kernel(const bf16* __restrict__ qkv, bf16* __restrict__ Qo,
                              bf16* __restrict__ Ko, bf16* __restrict__ Vo) {
    size_t n = (size_t)Bc * Hc * Tc * HDc;
    size_t i = (size_t)blockIdx.x * blockDim.x + threadIdx.x;
    size_t stride = (size_t)gridDim.x * blockDim.x;
    for (; i < n; i += stride) {
        int d = (int)(i & 63);
        size_t r = i >> 6;
        int t = (int)(r % Tc); r /= Tc;
        int h = (int)(r % Hc);
        int b = (int)(r / Hc);
        size_t base = ((size_t)(b * Tc + t)) * N3c + h * (3 * HDc);
        float kq = (float)qkv[base + d];
        float qq = (float)qkv[base + HDc + d];
        float vq = (float)qkv[base + 2 * HDc + d];
        Qo[i] = (bf16)(qq * 0.125f); // fold HD^-0.5 into Q
        Vo[i] = (bf16)vq;
        Ko[(((size_t)(b * Hc + h)) * HDc + d) * Tc + t] = (bf16)kq;
    }
}

// ---------------------------------------------------------------- flash attention
// grid: (T/32, B*H); block 256. Each block: 32 queries x all keys, causal.
__global__ __launch_bounds__(256) void attention_kernel(const bf16* __restrict__ Q,
                                                        const bf16* __restrict__ K,
                                                        const bf16* __restrict__ V,
                                                        bf16* __restrict__ O) {
    __shared__ float Ssc[32][256];
    __shared__ __align__(16) bf16 Pbf[32][256];
    __shared__ float m_s[32], l_s[32], f_s[32];

    const int tid = threadIdx.x, lane = tid & 31;
    // wave-uniform index -> SGPR, so tile-skip guards are scalar branches and
    // WMMA always executes with EXEC all-ones (ISA requirement).
    const int wid = __builtin_amdgcn_readfirstlane(tid >> 5);
    const int qtile = blockIdx.x;
    const int bh = blockIdx.y;
    const int b = bh / Hc, h = bh % Hc;
    const bf16* Qp = Q + (size_t)bh * Tc * HDc;
    const bf16* Kp = K + (size_t)bh * HDc * Tc;
    const bf16* Vp = V + (size_t)bh * Tc * HDc;

    const int qmax = qtile * 32 + 31;
    const int oi = wid >> 2, oj = wid & 3; // wave's O tile: q-half, d-quarter
    v8f acc = {};

    if (tid < 32) { m_s[tid] = -INFINITY; l_s[tid] = 0.f; }
    __syncthreads();

    const int mlane = lane & 15;
    const int khalf = lane >> 4;
    const int mhalf = lane >> 4;

    const int nchunk = (qtile >> 3) + 1;
    for (int c = 0; c < nchunk; ++c) {
        const int k0g = c << 8;
        // ---- S = (Q*scale) @ K^T for this 32x256 block (wave-uniform tiles)
#pragma unroll
        for (int i = 0; i < 2; ++i) {
#pragma unroll
            for (int ktl = 0; ktl < 2; ++ktl) {
                const int kt = wid * 2 + ktl;      // SGPR
                const int keybase = k0g + kt * 16; // SGPR
                if (keybase > qmax) continue;      // scalar branch, EXEC untouched
                v8f s = {};
#pragma unroll
                for (int kk = 0; kk < HDc; kk += 32) {
                    const int m = qtile * 32 + i * 16 + mlane;
                    const u32x4* alo = (const u32x4*)(Qp + (size_t)m * HDc + kk + khalf * 8);
                    const u32x4* ahi = (const u32x4*)(Qp + (size_t)m * HDc + kk + 16 + khalf * 8);
                    v16bf a = make_frag(*alo, *ahi);
                    v16bf bb;
#pragma unroll
                    for (int s = 0; s < 16; ++s) {
                        int k = (s < 8) ? (khalf * 8 + s) : (16 + khalf * 8 + (s - 8));
                        bb[s] = Kp[(size_t)(kk + k) * Tc + keybase + mlane];
                    }
                    s = WMMA_BF16(a, bb, s);
                }
#pragma unroll
                for (int r = 0; r < 8; ++r)
                    Ssc[i * 16 + mhalf * 8 + r][kt * 16 + mlane] = s[r];
            }
        }
        __syncthreads();
        // ---- online softmax (8 threads per query row)
        {
            const int row = tid >> 3, sub = tid & 7;
            const int qglob = qtile * 32 + row;
            const int col0 = sub * 32;
            float lm = -INFINITY;
            for (int cc = 0; cc < 32; ++cc) {
                int kg = k0g + col0 + cc;
                if (kg <= qglob) lm = fmaxf(lm, Ssc[row][col0 + cc]);
            }
            for (int off = 1; off < 8; off <<= 1)
                lm = fmaxf(lm, __shfl_xor(lm, off, 32));
            float mold = m_s[row];
            float mnew = fmaxf(mold, lm);
            float fac  = __expf(mold - mnew);
            float psum = 0.f;
            for (int cc = 0; cc < 32; ++cc) {
                int cl = col0 + cc;
                int kg = k0g + cl;
                float p = (kg <= qglob) ? __expf(Ssc[row][cl] - mnew) : 0.f;
                Pbf[row][cl] = (bf16)p;
                psum += p;
            }
            for (int off = 1; off < 8; off <<= 1)
                psum += __shfl_xor(psum, off, 32);
            if (sub == 0) {
                m_s[row] = mnew;
                f_s[row] = fac;
                l_s[row] = l_s[row] * fac + psum;
            }
        }
        __syncthreads();
        // ---- rescale running O, then O += P @ V
        {
#pragma unroll
            for (int r = 0; r < 8; ++r)
                acc[r] *= f_s[oi * 16 + mhalf * 8 + r];
#pragma unroll
            for (int kk = 0; kk < 256; kk += 32) {
                if (k0g + kk > qmax) break; // block-uniform scalar branch
                const int m = oi * 16 + mlane;
                const u32x4* alo = (const u32x4*)(&Pbf[m][kk + khalf * 8]);
                const u32x4* ahi = (const u32x4*)(&Pbf[m][kk + 16 + khalf * 8]);
                v16bf a = make_frag(*alo, *ahi);
                v16bf bb;
#pragma unroll
                for (int s = 0; s < 16; ++s) {
                    int k = (s < 8) ? (khalf * 8 + s) : (16 + khalf * 8 + (s - 8));
                    bb[s] = Vp[(size_t)(k0g + kk + k) * HDc + oj * 16 + mlane];
                }
                acc = WMMA_BF16(a, bb, acc);
            }
        }
    }
    // ---- normalize and scatter to [B,T,H*HD]
    {
        const int n = lane & 15;
#pragma unroll
        for (int r = 0; r < 8; ++r) {
            int row = oi * 16 + mhalf * 8 + r;
            int qg  = qtile * 32 + row;
            float o = acc[r] / l_s[row];
            O[((size_t)(b * Tc + qg)) * Dc + h * HDc + oj * 16 + n] = (bf16)o;
        }
    }
}

// ---------------------------------------------------------------- NLL per row
__global__ __launch_bounds__(256) void nll_kernel(const float* __restrict__ logits,
                                                  const int* __restrict__ tgt,
                                                  float* __restrict__ row_nll, int Vn) {
    __shared__ float red[256];
    const int row = blockIdx.x, tid = threadIdx.x;
    const float* lr = logits + (size_t)row * Vn;
    float mx = -INFINITY;
    for (int j = tid; j < Vn; j += 256) mx = fmaxf(mx, lr[j]);
    red[tid] = mx; __syncthreads();
    for (int off = 128; off > 0; off >>= 1) {
        if (tid < off) red[tid] = fmaxf(red[tid], red[tid + off]);
        __syncthreads();
    }
    mx = red[0]; __syncthreads();
    float s = 0.f;
    for (int j = tid; j < Vn; j += 256) s += __expf(lr[j] - mx);
    red[tid] = s; __syncthreads();
    for (int off = 128; off > 0; off >>= 1) {
        if (tid < off) red[tid] += red[tid + off];
        __syncthreads();
    }
    if (tid == 0) {
        float lse = logf(red[0]) + mx;
        row_nll[row] = lse - lr[tgt[row]];
    }
}

__global__ __launch_bounds__(256) void loss_reduce_kernel(const float* __restrict__ row_nll,
                                                          float* __restrict__ out, int n) {
    __shared__ float red[256];
    float s = 0.f;
    for (int i = threadIdx.x; i < n; i += 256) s += row_nll[i];
    red[threadIdx.x] = s; __syncthreads();
    for (int off = 128; off > 0; off >>= 1) {
        if (threadIdx.x < off) red[threadIdx.x] += red[threadIdx.x + off];
        __syncthreads();
    }
    if (threadIdx.x == 0) out[0] = red[0] / n;
}

// ---------------------------------------------------------------- launch
extern "C" void kernel_launch(void* const* d_in, const int* in_sizes, int n_in,
                              void* d_out, int out_size, void* d_ws, size_t ws_size,
                              hipStream_t stream) {
    (void)in_sizes; (void)n_in; (void)out_size; (void)ws_size;

    const int*   x_ids   = (const int*)  d_in[0];
    const int*   targets = (const int*)  d_in[1];
    const float* tok_emb = (const float*)d_in[2];
    const float* pos_emb = (const float*)d_in[3];
    const float* ln1_g   = (const float*)d_in[4];
    const float* ln1_b   = (const float*)d_in[5];
    const float* cproj_w = (const float*)d_in[6];
    const float* proj_w  = (const float*)d_in[7];
    const float* proj_b  = (const float*)d_in[8];
    const float* ln2_g   = (const float*)d_in[9];
    const float* ln2_b   = (const float*)d_in[10];
    const float* ffn_w1  = (const float*)d_in[11];
    const float* ffn_b1  = (const float*)d_in[12];
    const float* ffn_w2  = (const float*)d_in[13];
    const float* ffn_b2  = (const float*)d_in[14];
    const float* lnf_g   = (const float*)d_in[15];
    const float* lnf_b   = (const float*)d_in[16];
    const float* lm_w    = (const float*)d_in[17];
    const float* lm_b    = (const float*)d_in[18];

    char* ws = (char*)d_ws;
    size_t off = 0;
    auto alloc = [&](size_t bytes) -> void* {
        off = (off + 255) & ~(size_t)255;
        void* p = ws + off;
        off += bytes;
        return p;
    };

    // bf16 weight copies
    bf16* cprojW = (bf16*)alloc((size_t)Lc * Dc * N3c * 2);
    bf16* projW  = (bf16*)alloc((size_t)Lc * Dc * Dc * 2);
    bf16* ffn1W  = (bf16*)alloc((size_t)Lc * Dc * Fc * 2);
    bf16* ffn2W  = (bf16*)alloc((size_t)Lc * Fc * Dc * 2);
    bf16* lmW    = (bf16*)alloc((size_t)Dc * Vc * 2);
    // activations
    float* hA    = (float*)alloc((size_t)Mc * Dc * 4); // residual stream (block input)
    float* actB  = (float*)alloc((size_t)Mc * Dc * 4); // post-attention residual
    bf16*  lnD   = (bf16*) alloc((size_t)Mc * Dc * 2);
    bf16*  hid   = (bf16*) alloc((size_t)Mc * Fc * 2);
    bf16*  qkvB  = (bf16*) alloc((size_t)Mc * N3c * 2);
    bf16*  Qb    = (bf16*) alloc((size_t)Mc * Dc * 2);
    bf16*  Ktb   = (bf16*) alloc((size_t)Mc * Dc * 2);
    bf16*  Vb    = (bf16*) alloc((size_t)Mc * Dc * 2);
    bf16*  attnB = (bf16*) alloc((size_t)Mc * Dc * 2);
    float* rowNll= (float*)alloc((size_t)Mc * 4);

    float* logits = (float*)d_out;
    float* lossp  = (float*)d_out + (size_t)Mc * Vc;

    // ---- weight conversion (f32 -> bf16)
    cvt_kernel<<<2048, 256, 0, stream>>>(cproj_w, cprojW, (size_t)Lc * Dc * N3c);
    cvt_kernel<<<2048, 256, 0, stream>>>(proj_w,  projW,  (size_t)Lc * Dc * Dc);
    cvt_kernel<<<2048, 256, 0, stream>>>(ffn_w1,  ffn1W,  (size_t)Lc * Dc * Fc);
    cvt_kernel<<<2048, 256, 0, stream>>>(ffn_w2,  ffn2W,  (size_t)Lc * Fc * Dc);
    cvt_kernel<<<2048, 256, 0, stream>>>(lm_w,    lmW,    (size_t)Dc * Vc);

    // ---- embedding
    embed_kernel<<<2048, 256, 0, stream>>>(x_ids, tok_emb, pos_emb, hA);

    const dim3 blk(256);
    const dim3 gQKV(N3c / 64, Mc / 128);
    const dim3 gD(Dc / 64, Mc / 128);
    const dim3 gF(Fc / 64, Mc / 128);
    const dim3 gLM((Vc + 63) / 64, Mc / 128);
    const dim3 gAttn(Tc / 32, Bc * Hc);

    for (int l = 0; l < Lc; ++l) {
        // ln1 -> bf16
        layernorm_kernel<<<Mc, blk, 0, stream>>>(hA, ln1_g + (size_t)l * Dc,
                                                 ln1_b + (size_t)l * Dc, lnD, Dc);
        // qkv = ln1 @ cproj_w  (no bias)
        gemm_bf16_kernel<true><<<gQKV, blk, 0, stream>>>(
            lnD, cprojW + (size_t)l * Dc * N3c, nullptr, nullptr,
            nullptr, qkvB, Mc, N3c, Dc, 0);
        // split k|q|v, scale q, transpose k
        repack_kernel<<<4096, blk, 0, stream>>>(qkvB, Qb, Ktb, Vb);
        // causal flash attention
        attention_kernel<<<gAttn, blk, 0, stream>>>(Qb, Ktb, Vb, attnB);
        // act = hA + attn @ proj_w + proj_b
        gemm_bf16_kernel<true><<<gD, blk, 0, stream>>>(
            attnB, projW + (size_t)l * Dc * Dc, proj_b + (size_t)l * Dc, hA,
            actB, nullptr, Mc, Dc, Dc, 0);
        // ln2 -> bf16
        layernorm_kernel<<<Mc, blk, 0, stream>>>(actB, ln2_g + (size_t)l * Dc,
                                                 ln2_b + (size_t)l * Dc, lnD, Dc);
        // hid = relu(ln2 @ w1 + b1)
        gemm_bf16_kernel<true><<<gF, blk, 0, stream>>>(
            lnD, ffn1W + (size_t)l * Dc * Fc, ffn_b1 + (size_t)l * Fc, nullptr,
            nullptr, hid, Mc, Fc, Dc, 1);
        // hA = hA + hid @ w2 + b2   (NOTE: residual is the BLOCK INPUT, per reference)
        gemm_bf16_kernel<true><<<gD, blk, 0, stream>>>(
            hid, ffn2W + (size_t)l * Fc * Dc, ffn_b2 + (size_t)l * Dc, hA,
            hA, nullptr, Mc, Dc, Fc, 0);
    }

    // final LN + LM head
    layernorm_kernel<<<Mc, blk, 0, stream>>>(hA, lnf_g, lnf_b, lnD, Dc);
    gemm_bf16_kernel<false><<<gLM, blk, 0, stream>>>(
        lnD, lmW, lm_b, nullptr, logits, nullptr, Mc, Vc, Dc, 0);

    // loss = mean(-log_softmax(logits)[targets])
    nll_kernel<<<Mc, blk, 0, stream>>>(logits, targets, rowNll, Vc);
    loss_reduce_kernel<<<1, blk, 0, stream>>>(rowNll, lossp, Mc);
}
